// LSTMTheta_16973710754334
// MI455X (gfx1250) — matrix operand order repaired and checked by reading, not compiled, not run on hardware
//
#include <hip/hip_runtime.h>
#include <hip/hip_bf16.h>
#include <stdint.h>

// Problem constants (from reference): B=256, T=128, IN=512, H=1024, T_PRED=32
#define NB   256
#define NT   128
#define NIN  512
#define NH   1024
#define NTP  32
#define N4H  4096           // 4*H
#define KCAT 1536           // IN + H

typedef __attribute__((ext_vector_type(16))) __bf16 bf16x16;
typedef __attribute__((ext_vector_type(8)))  __bf16 bf16x8;
typedef __attribute__((ext_vector_type(8)))  float  f32x8;

// ---------------------------------------------------------------------------
// Fragment loaders for v_wmma_f32_16x16x32_bf16; A and B both use contiguous k order (k = 16*half + s).
//
// A (16x32, row-major in memory, row = lane&15):
//   lanes 0-15 : elems 0..15 = K k0+0..15
//   lanes 16-31: elems 0..15 = K k0+16..31
// => per lane: 16 contiguous bf16 (two 16B loads) at k0 + 16*half.
//
// B (32x16, stored column-major: column n contiguous over K, stride ldb):
//   lanes 0-15 hold K k0..k0+15 of column lane; lanes 16-31 hold K k0+16..k0+31.
// => per lane: 16 contiguous bf16 at column base + k0 + 16*half.
// ---------------------------------------------------------------------------
__device__ __forceinline__ bf16x16 load_a_frag(const __bf16* __restrict__ row,
                                               int k0) {
  const __bf16* p = row + k0;     // row pointer already includes half*16
  bf16x8 lo = *(const bf16x8*)(p);
  bf16x8 hi = *(const bf16x8*)(p + 8);
  return __builtin_shufflevector(lo, hi, 0, 1, 2, 3, 4, 5, 6, 7,
                                          8, 9, 10, 11, 12, 13, 14, 15);
}

__device__ __forceinline__ bf16x16 load_b_frag(const __bf16* __restrict__ col,
                                               int k0) {
  const __bf16* p = col + k0;     // col pointer already includes half*16
  bf16x8 lo = *(const bf16x8*)(p);
  bf16x8 hi = *(const bf16x8*)(p + 8);
  return __builtin_shufflevector(lo, hi, 0, 1, 2, 3, 4, 5, 6, 7,
                                          8, 9, 10, 11, 12, 13, 14, 15);
}

struct Frags {
  bf16x16 a, b0, b1, b2, b3;
};

__device__ __forceinline__ Frags load_frags(const __bf16* __restrict__ arow, int ka,
                                            const __bf16* __restrict__ c0,
                                            const __bf16* __restrict__ c1,
                                            const __bf16* __restrict__ c2,
                                            const __bf16* __restrict__ c3, int kb) {
  Frags f;
  f.a  = load_a_frag(arow, ka);
  f.b0 = load_b_frag(c0, kb);
  f.b1 = load_b_frag(c1, kb);
  f.b2 = load_b_frag(c2, kb);
  f.b3 = load_b_frag(c3, kb);
  return f;
}

__device__ __forceinline__ void mma4(const Frags& f, f32x8 acc[4]) {
  acc[0] = __builtin_amdgcn_wmma_f32_16x16x32_bf16(false, f.a, false, f.b0,
                                                   (short)0, acc[0], false, false);
  acc[1] = __builtin_amdgcn_wmma_f32_16x16x32_bf16(false, f.a, false, f.b1,
                                                   (short)0, acc[1], false, false);
  acc[2] = __builtin_amdgcn_wmma_f32_16x16x32_bf16(false, f.a, false, f.b2,
                                                   (short)0, acc[2], false, false);
  acc[3] = __builtin_amdgcn_wmma_f32_16x16x32_bf16(false, f.a, false, f.b3,
                                                   (short)0, acc[3], false, false);
}

// ---------------------------------------------------------------------------
// Generic bf16 WMMA GEMM with split-K A operand:
//   C[m][n] = sum_k A1[m][k]*B[k][n] (k<K1)  +  sum_k A2[m][k-K1]*B[k][n]
// B stored column-major (column n contiguous, leading dim ldb = K1+K2).
// Each 256-thread block: 2 (M) x 4 (N) waves; each wave -> 16x64 strip.
// Block tile = 32 (M) x 256 (N). grid.x = N/256, grid.y = M/32.
// Branch-free two-phase K loop, one-deep software pipeline (prefetch k+32
// fragments before issuing k's WMMAs so VMEM overlaps the matrix pipe).
// Optional fp32 + bf16 dual store; optional per-column bias.
// ---------------------------------------------------------------------------
__global__ __launch_bounds__(256)
void wmma_gemm_bf16(const __bf16* __restrict__ A1, int lda1, int K1,
                    const __bf16* __restrict__ A2, int lda2, int K2,
                    const __bf16* __restrict__ Bm, int ldb,
                    const float*  __restrict__ bias,
                    float*  __restrict__ Cout, int ldc,
                    __bf16* __restrict__ Cbf,  int ldcbf) {
  const int lane  = threadIdx.x & 31;
  const int wave  = threadIdx.x >> 5;
  const int waveM = wave >> 2;        // 0..1
  const int waveN = wave & 3;         // 0..3
  const int half  = lane >> 4;        // 0..1
  const int l15   = lane & 15;

  const int tile_m = blockIdx.y * 32 + waveM * 16;
  const int n_base = blockIdx.x * 256 + waveN * 64;

  const int mrow = tile_m + l15;
  // Fold per-lane half-offsets into the base pointers once.
  const __bf16* a1 = A1 + (size_t)mrow * lda1 + half * 16;
  const __bf16* a2 = (K2 > 0) ? (A2 + (size_t)mrow * lda2 + half * 16) : a1;

  const __bf16* c0 = Bm + (size_t)(n_base +  0 + l15) * ldb + half * 16;
  const __bf16* c1 = Bm + (size_t)(n_base + 16 + l15) * ldb + half * 16;
  const __bf16* c2 = Bm + (size_t)(n_base + 32 + l15) * ldb + half * 16;
  const __bf16* c3 = Bm + (size_t)(n_base + 48 + l15) * ldb + half * 16;

  f32x8 acc[4] = {f32x8{}, f32x8{}, f32x8{}, f32x8{}};

  const int K = K1 + K2;

  // Prologue: fetch k0 = 0 (K1 >= 32 always in this workload).
  Frags cur = load_frags(a1, 0, c0, c1, c2, c3, 0);

  int k0 = 32;
  // Phase 1: A from A1 (branch-free).
  for (; k0 < K1; k0 += 32) {
    Frags nxt = load_frags(a1, k0, c0, c1, c2, c3, k0);
    mma4(cur, acc);
    cur = nxt;
  }
  // Phase 2: A from A2 (branch-free).
  for (; k0 < K; k0 += 32) {
    Frags nxt = load_frags(a2, k0 - K1, c0, c1, c2, c3, k0);
    mma4(cur, acc);
    cur = nxt;
  }
  // Drain.
  mma4(cur, acc);

  // Epilogue: C layout -> lane n = n_base + 16t + l15 ; VGPR r -> row r + 8*half.
#pragma unroll
  for (int t = 0; t < 4; ++t) {
    const int n = n_base + t * 16 + l15;
    const float bn = bias ? bias[n] : 0.0f;
#pragma unroll
    for (int r = 0; r < 8; ++r) {
      const int m = tile_m + r + 8 * half;
      const float v = acc[t][r] + bn;
      Cout[(size_t)m * ldc + n] = v;
      if (Cbf) Cbf[(size_t)m * ldcbf + n] = (__bf16)v;
    }
  }
}

// ---------------------------------------------------------------------------
// Pointwise LSTM cell update. G = [256][4096] pre-activation gates (i,f,g,o).
// Updates c in place (fp32) and writes h as bf16 (feeds next step's GEMM).
// ---------------------------------------------------------------------------
__device__ __forceinline__ float sigm(float x) { return 1.0f / (1.0f + __expf(-x)); }
__device__ __forceinline__ float tanh_fast(float x) {
  float e = __expf(2.0f * x);
  return (e - 1.0f) / (e + 1.0f);
}

__global__ __launch_bounds__(256)
void lstm_pointwise(const float* __restrict__ G, float* __restrict__ c,
                    __bf16* __restrict__ hbf) {
  const int idx = blockIdx.x * 256 + threadIdx.x;   // 0 .. 256*1024-1
  const int b = idx >> 10;
  const int j = idx & (NH - 1);
  const float* g = G + (size_t)b * N4H;
  const float gi = sigm(g[j]);
  const float gf = sigm(g[j + NH]);
  const float gg = tanh_fast(g[j + 2 * NH]);
  const float go = sigm(g[j + 3 * NH]);
  const float cn = gf * c[idx] + gi * gg;
  c[idx] = cn;
  hbf[idx] = (__bf16)(go * tanh_fast(cn));
}

// ---------------------------------------------------------------------------
// One-time prep kernels
// ---------------------------------------------------------------------------
__global__ __launch_bounds__(256)
void cvt_f32_to_bf16(const float* __restrict__ src, __bf16* __restrict__ dst) {
  const int i = blockIdx.x * 256 + threadIdx.x;
  dst[i] = (__bf16)src[i];
}

// Wcat[n][k] (k contiguous, ld=1536): k<512 -> W_ih[n][k], else W_hh[n][k-512]
__global__ __launch_bounds__(256)
void build_wcat(const float* __restrict__ Wih, const float* __restrict__ Whh,
                __bf16* __restrict__ Wcat) {
  const int n = blockIdx.y;
  const int k = blockIdx.x * 256 + threadIdx.x;
  const float v = (k < NIN) ? Wih[(size_t)n * NIN + k]
                            : Whh[(size_t)n * NH + (k - NIN)];
  Wcat[(size_t)n * KCAT + k] = (__bf16)v;
}

// thT[n][k] = theta[k][n]  (theta: [H=1024][IN=512]); column-major B for pred.
__global__ __launch_bounds__(256)
void build_theta_t(const float* __restrict__ theta, __bf16* __restrict__ thT) {
  const int n = blockIdx.y;                        // 0..511
  const int k = blockIdx.x * 256 + threadIdx.x;    // 0..1023
  thT[(size_t)n * NH + k] = (__bf16)theta[(size_t)k * NIN + n];
}

__global__ __launch_bounds__(256)
void init_state(float* __restrict__ c, __bf16* __restrict__ hbf,
                const float* __restrict__ bih, const float* __restrict__ bhh,
                float* __restrict__ bsum) {
  const int i = blockIdx.x * 256 + threadIdx.x;    // 256*1024 threads
  c[i] = 0.0f;
  hbf[i] = (__bf16)0.0f;
  if (i < N4H) bsum[i] = bih[i] + bhh[i];
}

// ---------------------------------------------------------------------------
// Host driver
// ---------------------------------------------------------------------------
extern "C" void kernel_launch(void* const* d_in, const int* in_sizes, int n_in,
                              void* d_out, int out_size, void* d_ws, size_t ws_size,
                              hipStream_t stream) {
  const float* x     = (const float*)d_in[0];   // [256,128,512]
  const float* theta = (const float*)d_in[1];   // [1024,512]
  const float* W_ih  = (const float*)d_in[2];   // [4096,512]
  const float* W_hh  = (const float*)d_in[3];   // [4096,1024]
  const float* b_ih  = (const float*)d_in[4];   // [4096]
  const float* b_hh  = (const float*)d_in[5];   // [4096]
  float* out = (float*)d_out;                   // [256,32,512]

  // Workspace carve-up (all offsets 256B-aligned by construction)
  char* w = (char*)d_ws;
  __bf16* xbf   = (__bf16*)(w);                         // 256*128*512*2 = 33,554,432
  __bf16* Wcat  = (__bf16*)(w + 33554432);              // 4096*1536*2   = 12,582,912
  __bf16* thT   = (__bf16*)(w + 46137344);              // 512*1024*2    =  1,048,576
  float*  bsum  = (float* )(w + 47185920);              // 4096*4        =     16,384
  __bf16* hbf   = (__bf16*)(w + 47202304);              // 256*1024*2    =    524,288
  float*  cst   = (float* )(w + 47726592);              // 256*1024*4    =  1,048,576
  float*  G     = (float* )(w + 48775168);              // 256*4096*4    =  4,194,304
  __bf16* pbf   = (__bf16*)(w + 52969472);              // 256*512*2     =    262,144
  // total ws usage ~53.2 MB

  // --- one-time prep ---
  cvt_f32_to_bf16<<<(NB * NT * NIN) / 256, 256, 0, stream>>>(x, xbf);
  build_wcat<<<dim3(KCAT / 256, N4H), 256, 0, stream>>>(W_ih, W_hh, Wcat);
  build_theta_t<<<dim3(NH / 256, NIN), 256, 0, stream>>>(theta, thT);
  init_state<<<(NB * NH) / 256, 256, 0, stream>>>(cst, hbf, b_ih, b_hh, bsum);

  const dim3 gate_grid(N4H / 256, NB / 32);   // (16, 8)
  const dim3 pred_grid(NIN / 256, NB / 32);   // (2, 8)
  const int  ldo = NTP * NIN;                 // output row stride = 16384

  // --- encode: 128 recurrent steps ---
  for (int t = 0; t < NT; ++t) {
    // gates = x_t @ W_ih^T + h @ W_hh^T + b  ;  A1 = x[:, t, :], A2 = h
    wmma_gemm_bf16<<<gate_grid, 256, 0, stream>>>(
        xbf + (size_t)t * NIN, NT * NIN, NIN,
        hbf, NH, NH,
        Wcat, KCAT, bsum,
        G, N4H, (__bf16*)nullptr, 0);
    lstm_pointwise<<<(NB * NH) / 256, 256, 0, stream>>>(G, cst, hbf);
  }

  // --- decode: pred0 then 31 autoregressive steps ---
  // pred = h @ theta  -> out[:, t, :] (fp32) and pbf (bf16 feedback)
  wmma_gemm_bf16<<<pred_grid, 256, 0, stream>>>(
      hbf, NH, NH,
      (const __bf16*)nullptr, 0, 0,
      thT, NH, (const float*)nullptr,
      out, ldo, pbf, NIN);

  for (int t = 1; t < NTP; ++t) {
    wmma_gemm_bf16<<<gate_grid, 256, 0, stream>>>(
        pbf, NIN, NIN,
        hbf, NH, NH,
        Wcat, KCAT, bsum,
        G, N4H, (__bf16*)nullptr, 0);
    lstm_pointwise<<<(NB * NH) / 256, 256, 0, stream>>>(G, cst, hbf);
    wmma_gemm_bf16<<<pred_grid, 256, 0, stream>>>(
        hbf, NH, NH,
        (const __bf16*)nullptr, 0, 0,
        thT, NH, (const float*)nullptr,
        out + (size_t)t * NIN, ldo, pbf, NIN);
  }
}
